// VQCodebook_69930657513642
// MI455X (gfx1250) — matrix operand order repaired and checked by reading, not compiled, not run on hardware
//
#include <hip/hip_runtime.h>

typedef __attribute__((ext_vector_type(16))) __bf16 v16bf;
typedef __attribute__((ext_vector_type(8)))  float  v8f;

#define NB      32
#define DIM     64
#define HW      4096                 // 64*64
#define NROWS   (NB * HW)            // 131072
#define KCODES  1024
#define NCHUNK  (KCODES / 16)        // 64 chunks of 16 codes

// ---------------------------------------------------------------------------
// Pack codebook into the WMMA B-operand lane layout, split into bf16 hi/lo.
// For chunk j, K-group g (g=0: K 0..31, g=1: K 32..63), lane l:
//   16 contiguous bf16 = code (j*16 + l%16), dims g*32 + (l/16)*8 + 0..7, then g*32 + 16 + (l/16)*8 + 0..7
// ---------------------------------------------------------------------------
__global__ void vq_pack_cb(const float* __restrict__ cb,
                           __bf16* __restrict__ hi, __bf16* __restrict__ lo) {
  int t = blockIdx.x * blockDim.x + threadIdx.x;   // [0, NCHUNK*2*32)
  if (t >= NCHUNK * 2 * 32) return;
  int l = t & 31;
  int g = (t >> 5) & 1;
  int j = t >> 6;
  int code = j * 16 + (l & 15);
  int dim0 = g * 32 + ((l >> 4) << 3);
  const float* src = cb + code * DIM + dim0;
  __bf16* ph = hi + t * 16;
  __bf16* pl = lo + t * 16;
#pragma unroll
  for (int e = 0; e < 16; ++e) {
    float v  = src[e + (e & 8)];
    __bf16 vh = (__bf16)v;
    ph[e] = vh;
    pl[e] = (__bf16)(v - (float)vh);
  }
}

__global__ void vq_cnorm(const float* __restrict__ cb, float* __restrict__ cn) {
  int k = blockIdx.x * blockDim.x + threadIdx.x;
  if (k >= KCODES) return;
  float s = 0.f;
#pragma unroll 8
  for (int d = 0; d < DIM; ++d) { float v = cb[k * DIM + d]; s += v * v; }
  cn[k] = s;
}

// ---------------------------------------------------------------------------
// Main fused distance + argmin kernel. One wave per 16-row tile.
// A operand (16x32 bf16, ISA 7.12.2): lane l holds row M = l%16,
//   elems 0..7  -> K = (l/16)*8 + 0..7
//   elems 8..15 -> K = 16 + (l/16)*8 + 0..7
// C/D (16x16 f32): VGPR r, lanes 0..15 -> M=r, N=lane; lanes 16..31 -> M=8+r.
//
// Single accumulator chain + immediate epilogue: the 4-coexec WMMA->VALU
// hazard NOPs only stall this wave, not the SIMD's XDL pipe, and at ~10
// waves/SIMD they are fully hidden. (A deferred/ping-pong epilogue was tried
// and rejected: it raised register pressure enough that the allocator
// rematerialized the A-operand bf16 conversions inside the loop.)
// ---------------------------------------------------------------------------
__global__ __launch_bounds__(256) void vq_argmin(
    const float*  __restrict__ h,
    const __bf16* __restrict__ cbh,
    const __bf16* __restrict__ cbl,
    const float*  __restrict__ cnorm,
    int*          __restrict__ z) {
  const int lane = threadIdx.x & 31;
  const int wave = threadIdx.x >> 5;
  const int tile = blockIdx.x * 8 + wave;
  const int row  = tile * 16 + (lane & 15);     // flat query row for this lane
  const int kb   = (lane >> 4) * 8;             // K sub-base from lane half

  // flat[row][d] = h[(row/HW)*DIM*HW + d*HW + row%HW]
  const float* xbase = h + (size_t)(row >> 12) * (DIM * HW) + (row & (HW - 1));

  v16bf a0h = {}, a0l = {}, a1h = {}, a1l = {};
#pragma unroll
  for (int e = 0; e < 8; ++e) {
    float v; __bf16 vh;
    v = xbase[(size_t)(kb + e)      * HW]; vh = (__bf16)v; a0h[e]     = vh; a0l[e]     = (__bf16)(v - (float)vh);
    v = xbase[(size_t)(16 + kb + e) * HW]; vh = (__bf16)v; a0h[8 + e] = vh; a0l[8 + e] = (__bf16)(v - (float)vh);
    v = xbase[(size_t)(32 + kb + e) * HW]; vh = (__bf16)v; a1h[e]     = vh; a1l[e]     = (__bf16)(v - (float)vh);
    v = xbase[(size_t)(48 + kb + e) * HW]; vh = (__bf16)v; a1h[8 + e] = vh; a1l[8 + e] = (__bf16)(v - (float)vh);
  }

  const v16bf* bh = (const v16bf*)cbh;
  const v16bf* bl = (const v16bf*)cbl;

  float best[8]; int bidx[8];
#pragma unroll
  for (int r = 0; r < 8; ++r) { best[r] = 3.4e38f; bidx[r] = 0; }

#pragma unroll 1
  for (int j = 0; j < NCHUNK; ++j) {
    v16bf b0h = bh[(j * 2 + 0) * 32 + lane];
    v16bf b1h = bh[(j * 2 + 1) * 32 + lane];
    v16bf b0l = bl[(j * 2 + 0) * 32 + lane];
    v16bf b1l = bl[(j * 2 + 1) * 32 + lane];
    float cn  = cnorm[j * 16 + (lane & 15)];

    v8f acc = {};
    // x_hi . c_hi  (K = 0..63)
    acc = __builtin_amdgcn_wmma_f32_16x16x32_bf16(false, a0h, false, b0h, (short)0, acc, false, false);
    acc = __builtin_amdgcn_wmma_f32_16x16x32_bf16(false, a1h, false, b1h, (short)0, acc, false, false);
    // x_lo . c_hi
    acc = __builtin_amdgcn_wmma_f32_16x16x32_bf16(false, a0l, false, b0h, (short)0, acc, false, false);
    acc = __builtin_amdgcn_wmma_f32_16x16x32_bf16(false, a1l, false, b1h, (short)0, acc, false, false);
    // x_hi . c_lo
    acc = __builtin_amdgcn_wmma_f32_16x16x32_bf16(false, a0h, false, b0l, (short)0, acc, false, false);
    acc = __builtin_amdgcn_wmma_f32_16x16x32_bf16(false, a1h, false, b1l, (short)0, acc, false, false);

    const int cidx = j * 16 + (lane & 15);
#pragma unroll
    for (int r = 0; r < 8; ++r) {
      float dv = __builtin_fmaf(-2.0f, acc[r], cn);   // ||c||^2 - 2 x.c
      if (dv < best[r]) { best[r] = dv; bidx[r] = cidx; }
    }
  }

  // min-reduce across the 16 lanes of each half (N dimension), index tie-break
#pragma unroll
  for (int r = 0; r < 8; ++r) {
    float b = best[r]; int bi = bidx[r];
#pragma unroll
    for (int off = 8; off >= 1; off >>= 1) {          // stays within 16-lane half
      float ob = __shfl_xor(b, off, 32);
      int   oi = __shfl_xor(bi, off, 32);
      if (ob < b || (ob == b && oi < bi)) { b = ob; bi = oi; }
    }
    if ((lane & 15) == 0) {
      int m = tile * 16 + r + ((lane >> 4) << 3);     // lanes 0..15 -> M=r, 16..31 -> M=8+r
      z[m] = bi;
    }
  }
}

// ---------------------------------------------------------------------------
// q[n, :] = codebook[z[n], :]
// ---------------------------------------------------------------------------
__global__ void vq_gather(const int* __restrict__ z, const float* __restrict__ cb,
                          float* __restrict__ q) {
  int t = blockIdx.x * blockDim.x + threadIdx.x;
  int n = t >> 6;
  int d = t & 63;
  q[t] = cb[z[n] * DIM + d];
}

extern "C" void kernel_launch(void* const* d_in, const int* in_sizes, int n_in,
                              void* d_out, int out_size, void* d_ws, size_t ws_size,
                              hipStream_t stream) {
  const float* h  = (const float*)d_in[0];   // [32, 64, 64, 64] fp32
  const float* cb = (const float*)d_in[1];   // [1024, 64] fp32

  int*   z = (int*)d_out;                    // z: NROWS int32 tokens
  float* q = (float*)d_out + NROWS;          // q: NROWS*DIM fp32

  const int npack = NCHUNK * 2 * 32 * 16;    // 65536 bf16 elems per buffer
  __bf16* hi = (__bf16*)d_ws;
  __bf16* lo = hi + npack;
  float*  cn = (float*)(lo + npack);         // total ws use: 128KB+128KB+4KB

  vq_pack_cb<<<(NCHUNK * 2 * 32 + 255) / 256, 256, 0, stream>>>(cb, hi, lo);
  vq_cnorm<<<(KCODES + 255) / 256, 256, 0, stream>>>(cb, cn);
  vq_argmin<<<NROWS / 128, 256, 0, stream>>>(h, hi, lo, cn, z);
  vq_gather<<<(NROWS * DIM) / 256, 256, 0, stream>>>(z, cb, q);
}
